// ALiBiAttention_77790447665562
// MI455X (gfx1250) — matrix operand order repaired and checked by reading, not compiled, not run on hardware
//
#include <hip/hip_runtime.h>

#define BN   2
#define SEQ  2048
#define HIDN 1024
#define NH   16
#define HD   64
#define MROWS (BN * SEQ) /* 4096 */

#define LOG2E 1.4426950408889634f

typedef __attribute__((ext_vector_type(16))) __bf16 v16bf;
typedef __attribute__((ext_vector_type(8)))  __bf16 v8bf;
typedef __attribute__((ext_vector_type(8)))  float  v8f;
typedef int v4i_g __attribute__((vector_size(16)));   // matches builtin param type

static __device__ __forceinline__ v8f wmma_acc(v16bf a, v16bf b, v8f c) {
    // D = A(16x32 bf16) * B(32x16 bf16) + C(16x16 f32)
    return __builtin_amdgcn_wmma_f32_16x16x32_bf16(
        /*neg_a=*/false, a, /*neg_b=*/false, b,
        /*c_mod=*/(short)0, c, /*reuse_a=*/false, /*reuse_b=*/false);
}

// A-fragment (16x32, 16-bit): lane L holds row m=L%16; elem i<8 -> k = 8*(L/16)+i,
// elem i>=8 -> k = 16 + 8*(L/16) + (i-8).  off0 = kbase + 8*(L/16).
static __device__ __forceinline__ v16bf afrag_bf16(const __bf16* row, int off0) {
    v8bf x0 = *(const v8bf*)(row + off0);
    v8bf x1 = *(const v8bf*)(row + off0 + 16);
    v16bf a;
#pragma unroll
    for (int i = 0; i < 8; ++i) { a[i] = x0[i]; a[i + 8] = x1[i]; }
    return a;
}

static __device__ __forceinline__ v16bf afrag_f32(const float* row, int off0) {
    v8f x0 = *(const v8f*)(row + off0);
    v8f x1 = *(const v8f*)(row + off0 + 16);
    v16bf a;
#pragma unroll
    for (int i = 0; i < 8; ++i) { a[i] = (__bf16)x0[i]; a[i + 8] = (__bf16)x1[i]; }
    return a;
}

// B-fragment (32x16, 16-bit): lane L holds col n=L%16; elem i<8 -> k = 8*(L/16)+i, elem i>=8 -> k = 16+8*(L/16)+(i-8).
static __device__ __forceinline__ v16bf bfrag_f32(const float* row, int off) {
    v8f x0 = *(const v8f*)(row + off);
    v8f x1 = *(const v8f*)(row + off + 16);
    v16bf b;
#pragma unroll
    for (int i = 0; i < 8; ++i) { b[i] = (__bf16)x0[i]; b[i + 8] = (__bf16)x1[i]; }
    return b;
}

// ---- gfx1250 async global->LDS copy (16 bytes / lane), ASYNCcnt-tracked ----
static __device__ __forceinline__ void async_cp_b128(void* lds_dst, const void* gsrc) {
#if defined(__has_builtin) && __has_builtin(__builtin_amdgcn_global_load_async_to_lds_b128)
    __builtin_amdgcn_global_load_async_to_lds_b128(
        (__attribute__((address_space(1))) v4i_g*)(uintptr_t)gsrc,
        (__attribute__((address_space(3))) v4i_g*)(unsigned)(uintptr_t)lds_dst,
        0, 0);
#else
    const unsigned l = (unsigned)(uintptr_t)lds_dst;  // generic LDS addr[31:0] = LDS offset
    asm volatile("global_load_async_to_lds_b128 %0, %1, off"
                 :: "v"(l), "v"(gsrc)
                 : "memory");
#endif
}

static __device__ __forceinline__ void wait_async0() {
#if defined(__has_builtin) && __has_builtin(__builtin_amdgcn_s_wait_asynccnt)
    __builtin_amdgcn_s_wait_asynccnt(0);
#else
    asm volatile("s_wait_asynccnt 0x0" ::: "memory");
#endif
}

// ---------------------------------------------------------------------------
// GEMM: Y[m][n] = sum_k A[m][k] * W[n][k] + bias[n]   (A row-major MxK, W NxK)
// MODE 0: bf16 out -> [B,H,S,D]     (Q / K)
// MODE 2: bf16 out -> [B,H,D,S]     (V transposed)
// MODE 3: A is bf16 (ctx), f32 out row-major [M][N]   (output projection)
// Block: 256 thr = 8 waves, 4(M) x 2(N); wave tile 32(M) x 64(N).
// ---------------------------------------------------------------------------
template <int MODE>
__global__ __launch_bounds__(256) void gemm_proj(const float* __restrict__ Af,
                                                 const __bf16* __restrict__ Ab,
                                                 const float* __restrict__ W,
                                                 const float* __restrict__ bias,
                                                 __bf16* __restrict__ outb,
                                                 float* __restrict__ outf) {
    const int lane = threadIdx.x & 31;
    const int wv   = threadIdx.x >> 5;
    const int ln16 = lane & 15;
    const int lh   = lane >> 4;   // 0 or 1

    const int mbase = blockIdx.x * 128 + (wv & 3) * 32;
    const int nbase = blockIdx.y * 128 + (wv >> 2) * 64;

    v8f acc[2][4] = {};
    const int r0 = mbase + ln16, r1 = mbase + 16 + ln16;
    const float* wp[4];
#pragma unroll
    for (int j = 0; j < 4; ++j) wp[j] = W + (size_t)(nbase + j * 16 + ln16) * HIDN;

    for (int k0 = 0; k0 < HIDN; k0 += 32) {
        v16bf a0, a1;
        if constexpr (MODE == 3) {
            a0 = afrag_bf16(Ab + (size_t)r0 * HIDN, k0 + 8 * lh);
            a1 = afrag_bf16(Ab + (size_t)r1 * HIDN, k0 + 8 * lh);
        } else {
            a0 = afrag_f32(Af + (size_t)r0 * HIDN, k0 + 8 * lh);
            a1 = afrag_f32(Af + (size_t)r1 * HIDN, k0 + 8 * lh);
        }
#pragma unroll
        for (int j = 0; j < 4; ++j) {
            const v16bf bj = bfrag_f32(wp[j], k0 + 8 * lh);
            acc[0][j] = wmma_acc(a0, bj, acc[0][j]);
            acc[1][j] = wmma_acc(a1, bj, acc[1][j]);
        }
    }

    // C/D layout: lane col n = L%16; VGPR v holds row m = v + 8*(L/16).
#pragma unroll
    for (int i = 0; i < 2; ++i) {
#pragma unroll
        for (int j = 0; j < 4; ++j) {
            const int n    = nbase + j * 16 + ln16;
            const float bn = bias[n];
#pragma unroll
            for (int v = 0; v < 8; ++v) {
                const int m   = mbase + i * 16 + v + 8 * lh;
                const float y = acc[i][j][v] + bn;
                if constexpr (MODE == 3) {
                    outf[(size_t)m * HIDN + n] = y;
                } else {
                    const int bb = m >> 11, s = m & (SEQ - 1);
                    const int hh = n >> 6,  d = n & (HD - 1);
                    if constexpr (MODE == 2)
                        outb[(((size_t)bb * NH + hh) * HD + d) * SEQ + s] = (__bf16)y;
                    else
                        outb[(((size_t)bb * NH + hh) * SEQ + s) * HD + d] = (__bf16)y;
                }
            }
        }
    }
}

// ---------------------------------------------------------------------------
// Flash attention with ALiBi, base-2 softmax. One wave = 32 queries of one
// (b,h); all 8 waves of a block share (b,h) and consume the SAME K/V stream,
// so K/V tiles are staged once per block into LDS with gfx1250 async
// global->LDS copies, double-buffered (prefetch tile n+1 while computing n).
// Scores computed TRANSPOSED (T = K_tile @ Q^T) so the score C-layout equals
// the A-fragment layout needed for P@V with zero cross-lane movement.
// ---------------------------------------------------------------------------
#define KSTRIDE 72  /* padded LDS row stride for K tile (elems) */
#define VSTRIDE 40  /* padded LDS row stride for V tile (elems) */

__global__ __launch_bounds__(256) void attn_kernel(const __bf16* __restrict__ Q,
                                                   const __bf16* __restrict__ K,
                                                   const __bf16* __restrict__ Vt,
                                                   const float* __restrict__ slopes,
                                                   __bf16* __restrict__ ctx) {
    __shared__ alignas(16) __bf16 sk[2][32 * KSTRIDE];
    __shared__ alignas(16) __bf16 sv[2][64 * VSTRIDE];

    const int tid  = threadIdx.x;
    const int lane = tid & 31;
    const int wv   = tid >> 5;
    const int ln16 = lane & 15;
    const int lh   = lane >> 4;

    const int task  = blockIdx.x * 8 + wv;   // 2048 tasks = 32 bh * 64 qtiles
    const int bh    = task >> 6;             // same for all waves in the block
    const int qt32  = task & 63;
    const int h     = bh & (NH - 1);
    const int qbase = qt32 * 32;
    const float slope2 = slopes[h] * LOG2E;
    const float scale2 = 0.125f * LOG2E;     // (1/sqrt(64)) * log2(e)

    const __bf16* kb = K  + (size_t)bh * SEQ * HD;
    const __bf16* vb = Vt + (size_t)bh * HD * SEQ;

    // Staging decomposition (256 threads): K tile 32x64 is contiguous 4 KB;
    // V tile is 64 rows x 32 keys with global row stride SEQ.
    const int kr = tid >> 3, ks = tid & 7;   // K: row 0..31, 16B segment 0..7
    const int vd = tid >> 2, vp = tid & 3;   // V: row 0..63, 16B segment 0..3

    // Q as B-operand: lane n = query; elems = dims in table order (8*lh+i, 16+8*lh+i).
    const __bf16* qrow0 = Q + ((size_t)bh * SEQ + qbase + ln16) * HD;
    const __bf16* qrow1 = qrow0 + (size_t)16 * HD;
    v16bf bq[2][2];
    bq[0][0] = afrag_bf16(qrow0, 8 * lh);
    bq[0][1] = afrag_bf16(qrow0, 32 + 8 * lh);
    bq[1][0] = afrag_bf16(qrow1, 8 * lh);
    bq[1][1] = afrag_bf16(qrow1, 32 + 8 * lh);

    float mi[2] = {-3.0e38f, -3.0e38f};
    float li[2] = {0.0f, 0.0f};
    v8f o[2][4] = {};

    // ALiBi bias, incremental: bias(kt,qt,v) = bjb + cofs[kt][qt] + vbias[v]
    float vbias[8];
#pragma unroll
    for (int v = 0; v < 8; ++v) vbias[v] = slope2 * (float)v;
    float cofs[2][2];
#pragma unroll
    for (int kt = 0; kt < 2; ++kt)
#pragma unroll
        for (int qt = 0; qt < 2; ++qt)
            cofs[kt][qt] = slope2 * (float)(kt * 16 + 8 * lh - qt * 16 - ln16);
    float bjb = slope2 * (float)(-qbase);
    const float djb = slope2 * 32.0f;

    // Prefetch first K/V tile into buffer 0.
    async_cp_b128(&sk[0][kr * KSTRIDE + ks * 8], kb + (size_t)(tid << 3));
    async_cp_b128(&sv[0][vd * VSTRIDE + vp * 8], vb + (size_t)vd * SEQ + vp * 8);

    int ib = 0;
    for (int jb = 0; jb < SEQ; jb += 32, ib ^= 1) {
        wait_async0();       // this wave's async stores to LDS landed
        __syncthreads();     // everyone's did; previous buffer free for reuse

        if (jb + 32 < SEQ) {
            const int jn = jb + 32;
            async_cp_b128(&sk[ib ^ 1][kr * KSTRIDE + ks * 8],
                          kb + (size_t)jn * HD + (tid << 3));
            async_cp_b128(&sv[ib ^ 1][vd * VSTRIDE + vp * 8],
                          vb + (size_t)vd * SEQ + jn + vp * 8);
        }

        const __bf16* skb = sk[ib];
        const __bf16* svb = sv[ib];

        v16bf ka[2][2];
        {
            const __bf16* r0 = skb + (size_t)ln16 * KSTRIDE;
            const __bf16* r1 = skb + (size_t)(16 + ln16) * KSTRIDE;
            ka[0][0] = afrag_bf16(r0, 8 * lh);
            ka[0][1] = afrag_bf16(r0, 32 + 8 * lh);
            ka[1][0] = afrag_bf16(r1, 8 * lh);
            ka[1][1] = afrag_bf16(r1, 32 + 8 * lh);
        }

        v8f t[2][2] = {};
#pragma unroll
        for (int kt = 0; kt < 2; ++kt)
#pragma unroll
            for (int qt = 0; qt < 2; ++qt) {
                t[kt][qt] = wmma_acc(ka[kt][0], bq[qt][0], t[kt][qt]);
                t[kt][qt] = wmma_acc(ka[kt][1], bq[qt][1], t[kt][qt]);
            }

        v16bf pa[2];
        float corr[2];
#pragma unroll
        for (int qt = 0; qt < 2; ++qt) {
            float tm = -3.0e38f;
#pragma unroll
            for (int kt = 0; kt < 2; ++kt) {
                const float base = bjb + cofs[kt][qt];
#pragma unroll
                for (int v = 0; v < 8; ++v) {
                    const float s = __builtin_fmaf(t[kt][qt][v], scale2, base + vbias[v]);
                    t[kt][qt][v] = s;
                    tm = fmaxf(tm, s);
                }
            }
            tm = fmaxf(tm, __shfl_xor(tm, 16));
            const float mnew = fmaxf(mi[qt], tm);
            corr[qt] = __builtin_amdgcn_exp2f(mi[qt] - mnew);
            float psum = 0.0f;
            v16bf p = {};
#pragma unroll
            for (int v = 0; v < 8; ++v) {
                const float p0 = __builtin_amdgcn_exp2f(t[0][qt][v] - mnew);
                const float p1 = __builtin_amdgcn_exp2f(t[1][qt][v] - mnew);
                psum += p0 + p1;
                p[v]     = (__bf16)p0;   // A-frag elem v   -> key 8*lh+v
                p[v + 8] = (__bf16)p1;   // A-frag elem v+8 -> key 16+8*lh+v
            }
            psum += __shfl_xor(psum, 16);
            li[qt] = li[qt] * corr[qt] + psum;
            mi[qt] = mnew;
            pa[qt] = p;
        }

        // Rescale O: row v+8*lh of tile qt belongs to query (v+8*lh) whose
        // corr lives in lane (v+8*lh).
#pragma unroll
        for (int qt = 0; qt < 2; ++qt)
#pragma unroll
            for (int v = 0; v < 8; ++v) {
                const float cf = __shfl(corr[qt], v + 8 * lh);
                o[qt][0][v] *= cf; o[qt][1][v] *= cf;
                o[qt][2][v] *= cf; o[qt][3][v] *= cf;
            }

#pragma unroll
        for (int dt = 0; dt < 4; ++dt) {
            // V B-operand: lane col = dim d; elems = keys in table order (8*lh+i, 16+8*lh+i).
            const v16bf vbf = afrag_bf16(svb + (size_t)(dt * 16 + ln16) * VSTRIDE, 8 * lh);
            o[0][dt] = wmma_acc(pa[0], vbf, o[0][dt]);
            o[1][dt] = wmma_acc(pa[1], vbf, o[1][dt]);
        }

        bjb += djb;
    }

    // Normalize and write ctx [MROWS][HIDN] bf16.
    const int b = bh >> 4;
#pragma unroll
    for (int qt = 0; qt < 2; ++qt)
#pragma unroll
        for (int v = 0; v < 8; ++v) {
            const float linv = 1.0f / __shfl(li[qt], v + 8 * lh);
            const int row = b * SEQ + qbase + qt * 16 + v + 8 * lh;
#pragma unroll
            for (int dt = 0; dt < 4; ++dt) {
                const int col = h * HD + dt * 16 + ln16;
                ctx[(size_t)row * HIDN + col] = (__bf16)(o[qt][dt][v] * linv);
            }
        }
}

// ---------------------------------------------------------------------------
extern "C" void kernel_launch(void* const* d_in, const int* in_sizes, int n_in,
                              void* d_out, int out_size, void* d_ws, size_t ws_size,
                              hipStream_t stream) {
    (void)in_sizes; (void)n_in; (void)out_size; (void)ws_size;
    const float* x      = (const float*)d_in[0];
    const float* Wq     = (const float*)d_in[1];
    const float* bq     = (const float*)d_in[2];
    const float* Wk     = (const float*)d_in[3];
    const float* bk     = (const float*)d_in[4];
    const float* Wv     = (const float*)d_in[5];
    const float* bv     = (const float*)d_in[6];
    const float* Wo     = (const float*)d_in[7];
    const float* bo     = (const float*)d_in[8];
    const float* slopes = (const float*)d_in[9];
    float* out = (float*)d_out;

    char* w = (char*)d_ws;
    const size_t bhsd = (size_t)BN * NH * SEQ * HD * sizeof(__bf16);  // 8 MB
    __bf16* q   = (__bf16*)w; w += bhsd;
    __bf16* k   = (__bf16*)w; w += bhsd;
    __bf16* vt  = (__bf16*)w; w += bhsd;
    __bf16* ctx = (__bf16*)w; w += (size_t)MROWS * HIDN * sizeof(__bf16);

    const dim3 blk(256);
    const dim3 gg(MROWS / 128, HIDN / 128);  // 32 x 8

    gemm_proj<0><<<gg, blk, 0, stream>>>(x, nullptr, Wq, bq, q,  nullptr);
    gemm_proj<0><<<gg, blk, 0, stream>>>(x, nullptr, Wk, bk, k,  nullptr);
    gemm_proj<2><<<gg, blk, 0, stream>>>(x, nullptr, Wv, bv, vt, nullptr);

    attn_kernel<<<dim3(2048 / 8), blk, 0, stream>>>(q, k, vt, slopes, ctx);

    gemm_proj<3><<<gg, blk, 0, stream>>>(nullptr, ctx, Wo, bo, nullptr, out);
}
